// HSTUAttention_73229192397471
// MI455X (gfx1250) — matrix operand order repaired and checked by reading, not compiled, not run on hardware
//
#include <hip/hip_runtime.h>

typedef _Float16 half_t;
typedef __attribute__((ext_vector_type(2)))  _Float16 h2;
typedef __attribute__((ext_vector_type(8)))  _Float16 v8h;
typedef __attribute__((ext_vector_type(16))) _Float16 v16h;
typedef __attribute__((ext_vector_type(8)))  float    v8f;
typedef __attribute__((ext_vector_type(4)))  int      v4i;

// Problem constants
#define BB 8
#define NN 1024
#define HH 512
#define NHD 8
#define HD 64
#define ROWS (BB * NN)      // 8192
#define QKVU_N (4 * HH)     // 2048

#ifndef __has_builtin
#define __has_builtin(x) 0
#endif
#if __has_builtin(__builtin_amdgcn_global_load_async_to_lds_b128)
#define HAVE_ASYNC_LDS 1
#else
#define HAVE_ASYNC_LDS 0
#endif

// ---------------- async / copy helpers ----------------

#if HAVE_ASYNC_LDS
__device__ __forceinline__ void cp16B_async(half_t* dst_lds,
                                            const half_t* src_g) {
  // signature (probe-confirmed): (v4i* gaddr, v4i* lds, int offset, int cpol)
  __builtin_amdgcn_global_load_async_to_lds_b128((v4i*)src_g, (v4i*)dst_lds, 0,
                                                 0);
}
#endif

// copy 32 bytes (16 halfs) global -> LDS
__device__ __forceinline__ void cp32B(half_t* dst_lds, const half_t* src_g) {
#if HAVE_ASYNC_LDS
  cp16B_async(dst_lds, src_g);
  cp16B_async(dst_lds + 8, src_g + 8);
#else
  v8h a = *(const v8h*)src_g;
  v8h b = *(const v8h*)(src_g + 8);
  v8h* d = (v8h*)dst_lds;
  d[0] = a;
  d[1] = b;
#endif
}

__device__ __forceinline__ void async_join() {
#if HAVE_ASYNC_LDS
#if __has_builtin(__builtin_amdgcn_s_wait_asynccnt)
  __builtin_amdgcn_s_wait_asynccnt(0);
#else
  asm volatile("s_wait_asynccnt 0x0" ::: "memory");
#endif
#endif
}

// ---------------- WMMA helpers ----------------

__device__ __forceinline__ v8f wmma16(v16h a, v16h b, v8f c) {
  // D = A(16x32 f16) * B(32x16 f16) + C(16x16 f32)
  return __builtin_amdgcn_wmma_f32_16x16x32_f16(false, a, false, b, (short)0, c,
                                                false, false);
}

// A fragment (16x32 f16) from LDS tile stored row-major [row][k], stride ld.
// Per ISA: lanes 0-15 M=lane, lanes 16-31 M=lane-16;
// VGPR v holds K = 16*(v>>2) + 8*(lane>>4) + 2*(v&3) (+1)
__device__ __forceinline__ v16h lds_a_frag(const half_t* lds, int ld, int row0,
                                           int k0, int lane) {
  const half_t* base = lds + (row0 + (lane & 15)) * ld + k0;
  int hk = (lane >> 4) << 3;
  v16h a;
#pragma unroll
  for (int v = 0; v < 8; ++v) {
    int k = ((v >> 2) << 4) + hk + ((v & 3) << 1);
    h2 p = *(const h2*)(base + k);
    a[2 * v]     = p[0];
    a[2 * v + 1] = p[1];
  }
  return a;
}

// B fragment (32x16 f16) from LDS tile stored [n][k] (K contiguous), stride ld.
// Per ISA (dense): lane N = lane&15,
// K = 16*(v>>2) + 8*(lane>>4) + 2*(v&3) (+1)  -> two runs of 8 halfs per lane.
__device__ __forceinline__ v16h lds_b_frag(const half_t* lds, int ld, int n0,
                                           int k0, int lane) {
  const half_t* p =
      lds + (n0 + (lane & 15)) * ld + k0 + ((lane >> 4) << 3);
  v8h lo = *(const v8h*)p, hi = *(const v8h*)(p + 16);
  return __builtin_shufflevector(lo, hi, 0, 1, 2, 3, 4, 5, 6, 7, 8, 9, 10, 11,
                                 12, 13, 14, 15);
}

__device__ __forceinline__ float silu_f(float x) {
  // fast: v_exp + v_rcp, no IEEE div expansion
  return x * __fdividef(1.0f, 1.0f + __expf(-x));
}

// ---------------- Kernel 0a: f32 -> f16 convert ----------------

__global__ void f32_to_f16_kernel(const float* __restrict__ src,
                                  half_t* __restrict__ dst, int n) {
  int i = blockIdx.x * blockDim.x + threadIdx.x;
  if (i < n) dst[i] = (half_t)src[i];
}

// ---------------- Kernel 0b: qkvu_proj [k][n] -> whT [n][k] f16 ------------

__global__ void w_transpose_kernel(const float* __restrict__ src,
                                   half_t* __restrict__ dst) {
  int i = blockIdx.x * blockDim.x + threadIdx.x;  // 512*2048 elements
  int k = i >> 11, n = i & (QKVU_N - 1);
  dst[(size_t)n * HH + k] = (half_t)src[i];
}

// ---------------- Kernel 1: layernorm(seqs) -> f16 ----------------

__global__ __launch_bounds__(256) void ln_kernel(const float* __restrict__ x,
                                                 half_t* __restrict__ out) {
  int row = blockIdx.x;
  int t = threadIdx.x;
  const float* xr = x + (size_t)row * HH;
  float a = xr[t], b = xr[t + 256];
  __shared__ float r1[256], r2[256];
  r1[t] = a + b;
  r2[t] = a * a + b * b;
  __syncthreads();
#pragma unroll
  for (int off = 128; off > 0; off >>= 1) {
    if (t < off) { r1[t] += r1[t + off]; r2[t] += r2[t + off]; }
    __syncthreads();
  }
  float mu = r1[0] * (1.0f / HH);
  float var = r2[0] * (1.0f / HH) - mu * mu;
  float rstd = rsqrtf(var + 1e-8f);
  half_t* o = out + (size_t)row * HH;
  o[t]       = (half_t)((a - mu) * rstd);
  o[t + 256] = (half_t)((b - mu) * rstd);
}

// ---------------- Kernel 2: qkvu = silu(xh @ wh), scatter to heads ----------
// Double-buffered (ping-pong) LDS pipeline: global loads for tile k+1 overlap
// the WMMA work on tile k. B comes from the pre-transposed whT [n][k], so both
// tiles are contiguous b128 copies (no in-loop transpose scatter).

__global__ __launch_bounds__(128) void gemm_qkvu_kernel(
    const half_t* __restrict__ xh, const half_t* __restrict__ whT,
    half_t* __restrict__ qh, half_t* __restrict__ kh, half_t* __restrict__ vth,
    float* __restrict__ u_ws) {
  __shared__ half_t sA[2][64 * 32];  // [m][k]
  __shared__ half_t sB[2][64 * 32];  // [n][k]
  int n0 = blockIdx.x * 64;
  int m0 = blockIdx.y * 64;
  int t = threadIdx.x;
  int lane = t & 31, wave = t >> 5;

  int r = t >> 1, c = (t & 1) * 16;
  const half_t* gA = xh + (size_t)(m0 + r) * HH + c;
  const half_t* gB = whT + (size_t)(n0 + r) * HH + c;

  v8h a0, a1, b0, b1;
  // prologue: fetch tile 0
  a0 = *(const v8h*)(gA);
  a1 = *(const v8h*)(gA + 8);
  b0 = *(const v8h*)(gB);
  b1 = *(const v8h*)(gB + 8);
  {
    v8h* s = (v8h*)(sA[0] + r * 32 + c);
    s[0] = a0; s[1] = a1;
    v8h* s2 = (v8h*)(sB[0] + r * 32 + c);
    s2[0] = b0; s2[1] = b1;
  }

  v8f acc[4] = {};
#pragma unroll 2
  for (int it = 0; it < 16; ++it) {
    int cur = it & 1;
    if (it < 15) {  // fetch tile it+1 into registers (overlaps compute)
      int k0 = (it + 1) * 32;
      a0 = *(const v8h*)(gA + k0);
      a1 = *(const v8h*)(gA + k0 + 8);
      b0 = *(const v8h*)(gB + k0);
      b1 = *(const v8h*)(gB + k0 + 8);
    }
    __syncthreads();  // buf[cur] writes visible; prior reads of buf[cur^1] done
    if (it < 15) {
      int nxt = cur ^ 1;
      v8h* s = (v8h*)(sA[nxt] + r * 32 + c);
      s[0] = a0; s[1] = a1;
      v8h* s2 = (v8h*)(sB[nxt] + r * 32 + c);
      s2[0] = b0; s2[1] = b1;
    }
    v16h af = lds_a_frag(sA[cur], 32, wave * 16, 0, lane);
#pragma unroll
    for (int tn = 0; tn < 4; ++tn) {
      v16h bf = lds_b_frag(sB[cur], 32, tn * 16, 0, lane);
      acc[tn] = wmma16(af, bf, acc[tn]);
    }
  }

  // Epilogue: silu, route to q/k ([b,h,n,d] f16), vT ([b,h,d,n] f16), u (f32)
  int hi = lane >> 4, col = lane & 15;
  int i_base = m0 + wave * 16;
#pragma unroll
  for (int tn = 0; tn < 4; ++tn) {
    int j = n0 + tn * 16 + col;
    int region = j >> 9;   // 0:q 1:k 2:v 3:u
    int jc = j & 511;
    int h = jc >> 6, d = jc & 63;
#pragma unroll
    for (int g = 0; g < 8; ++g) {
      int i = i_base + g + 8 * hi;
      float sv = silu_f(acc[tn][g]);
      int b = i >> 10, n = i & 1023;
      size_t bh = (size_t)b * NHD + h;
      if (region == 3) {
        u_ws[(size_t)i * HH + jc] = sv;
      } else if (region == 2) {
        vth[(bh * HD + d) * NN + n] = (half_t)sv;   // transposed: [bh][d][n]
      } else {
        size_t idx = (bh * NN + n) * HD + d;
        if (region == 0) qh[idx] = (half_t)sv;
        else             kh[idx] = (half_t)sv;
      }
    }
  }
}

// ---------------- Kernel 3: flash-style HSTU attention ----------------

__global__ __launch_bounds__(128) void attn_kernel(
    const half_t* __restrict__ qh, const half_t* __restrict__ kh,
    const half_t* __restrict__ vth, const int* __restrict__ action_ts,
    const float* __restrict__ ts_w, const float* __restrict__ pos_w,
    float* __restrict__ av_ws) {
  __shared__ half_t sQ[64 * 64];   // [n][d]
  __shared__ half_t sK[32 * 64];   // [m][d]   (B layout for scores: N=m, K=d)
  __shared__ half_t sVT[64 * 32];  // [d][m]   (B layout for av:     N=d, K=m)
  __shared__ half_t sP[64 * 32];   // attn [n][m]
  __shared__ int sTsn[64];
  __shared__ int sTsm[32];

  int bh = blockIdx.x;         // b*NH + h
  int b = bh >> 3, h = bh & 7;
  int n0 = blockIdx.y * 64;
  int t = threadIdx.x;
  int lane = t & 31, wave = t >> 5;

  const half_t* qbase = qh + (size_t)bh * NN * HD;
  const half_t* kbase = kh + (size_t)bh * NN * HD;
  const half_t* vtbase = vth + (size_t)bh * HD * NN;  // [d][n]
  const int* tsb = action_ts + b * NN;

  {  // Q tile 64x64
    int r = t >> 1, c0 = (t & 1) * 32;
    cp32B(sQ + r * 64 + c0, qbase + (size_t)(n0 + r) * HD + c0);
    cp32B(sQ + r * 64 + c0 + 16, qbase + (size_t)(n0 + r) * HD + c0 + 16);
  }
  if (t < 64) {
    int nx = n0 + t + 1;
    if (nx > NN - 1) nx = NN - 1;
    sTsn[t] = tsb[nx];  // ext[:,1:] timestamp
  }
  async_join();
  __syncthreads();

  v16h aq0 = lds_a_frag(sQ, 64, wave * 16, 0, lane);
  v16h aq1 = lds_a_frag(sQ, 64, wave * 16, 32, lane);

  v8f avacc[4] = {};
  int hi = lane >> 4, col = lane & 15;
  int mend = n0 + 64;

  for (int m0 = 0; m0 < mend; m0 += 32) {
    __syncthreads();
    {  // K tile 32x64 (natural layout)
      int r = t >> 2, c0 = (t & 3) * 16;
      cp32B(sK + r * 64 + c0, kbase + (size_t)(m0 + r) * HD + c0);
    }
    {  // V tile from transposed vth: rows d, cols m -> contiguous copy
      int d = t >> 1, seg = (t & 1) * 16;
      cp32B(sVT + d * 32 + seg, vtbase + (size_t)d * NN + m0 + seg);
    }
    if (t < 32) sTsm[t] = tsb[m0 + t];
    async_join();
    __syncthreads();

    // scores: this wave's 16 rows x 32 cols (2 tiles), K = HD = 64
    v8f sc[2] = {};
#pragma unroll
    for (int tm = 0; tm < 2; ++tm) {
      v16h b0 = lds_b_frag(sK, 64, tm * 16, 0, lane);
      v16h b1 = lds_b_frag(sK, 64, tm * 16, 32, lane);
      sc[tm] = wmma16(aq0, b0, sc[tm]);
      sc[tm] = wmma16(aq1, b1, sc[tm]);
    }

    // bias + silu/N + causal mask -> sP (each wave owns its 16 rows)
#pragma unroll
    for (int tm = 0; tm < 2; ++tm) {
      int mi = m0 + tm * 16 + col;
      int tsm = sTsm[tm * 16 + col];
#pragma unroll
      for (int g = 0; g < 8; ++g) {
        int nrow = wave * 16 + g + 8 * hi;
        int ni = n0 + nrow;
        float x = sc[tm][g];
        int diff = sTsn[nrow] - tsm;
        float ad = fabsf((float)diff);
        ad = fmaxf(ad, 1.0f);
        int bucket = (int)(__logf(ad) * (1.0f / 0.301f));
        bucket = bucket < 0 ? 0 : (bucket > 64 ? 64 : bucket);
        x += pos_w[mi - ni + (NN - 1)] + ts_w[bucket];
        float sv = silu_f(x) * (1.0f / (float)NN);
        if (mi > ni) sv = 0.0f;  // causal (tril) mask
        sP[nrow * 32 + tm * 16 + col] = (half_t)sv;
      }
    }

    // av += attn(16x32) @ v(32x64); same-wave LDS round trip (DS in order)
    v16h ap = lds_a_frag(sP, 32, wave * 16, 0, lane);
#pragma unroll
    for (int td = 0; td < 4; ++td) {
      v16h bv = lds_b_frag(sVT, 32, td * 16, 0, lane);
      avacc[td] = wmma16(ap, bv, avacc[td]);
    }
  }

  // write av back as [b][n][h*64+d] f32
#pragma unroll
  for (int td = 0; td < 4; ++td) {
#pragma unroll
    for (int g = 0; g < 8; ++g) {
      int n = n0 + wave * 16 + g + 8 * hi;
      av_ws[((size_t)b * NN + n) * HH + h * HD + td * 16 + col] = avacc[td][g];
    }
  }
}

// ---------------- Kernel 4: y = u * layernorm(av) -> f16 ----------------

__global__ __launch_bounds__(256) void lnu_kernel(const float* __restrict__ av,
                                                  const float* __restrict__ u,
                                                  half_t* __restrict__ yh) {
  int row = blockIdx.x;
  int t = threadIdx.x;
  const float* xr = av + (size_t)row * HH;
  float a = xr[t], b = xr[t + 256];
  __shared__ float r1[256], r2[256];
  r1[t] = a + b;
  r2[t] = a * a + b * b;
  __syncthreads();
#pragma unroll
  for (int off = 128; off > 0; off >>= 1) {
    if (t < off) { r1[t] += r1[t + off]; r2[t] += r2[t + off]; }
    __syncthreads();
  }
  float mu = r1[0] * (1.0f / HH);
  float var = r2[0] * (1.0f / HH) - mu * mu;
  float rstd = rsqrtf(var + 1e-8f);
  const float* ur = u + (size_t)row * HH;
  half_t* o = yh + (size_t)row * HH;
  o[t]       = (half_t)(ur[t] * (a - mu) * rstd);
  o[t + 256] = (half_t)(ur[t + 256] * (b - mu) * rstd);
}

// ---------------- Kernel 5: out = yh @ out_w^T + out_b ----------------

__global__ __launch_bounds__(128) void gemm_out_kernel(
    const half_t* __restrict__ yh, const half_t* __restrict__ owh,
    const float* __restrict__ out_b, float* __restrict__ out) {
  __shared__ half_t sA[2][64 * 32];  // [m][k]
  __shared__ half_t sB[2][64 * 32];  // [n][k]
  int n0 = blockIdx.x * 64;
  int m0 = blockIdx.y * 64;
  int t = threadIdx.x;
  int lane = t & 31, wave = t >> 5;

  int r = t >> 1, c = (t & 1) * 16;
  const half_t* gA = yh + (size_t)(m0 + r) * HH + c;
  const half_t* gB = owh + (size_t)(n0 + r) * HH + c;

  v8h a0, a1, b0, b1;
  a0 = *(const v8h*)(gA);
  a1 = *(const v8h*)(gA + 8);
  b0 = *(const v8h*)(gB);
  b1 = *(const v8h*)(gB + 8);
  {
    v8h* s = (v8h*)(sA[0] + r * 32 + c);
    s[0] = a0; s[1] = a1;
    v8h* s2 = (v8h*)(sB[0] + r * 32 + c);
    s2[0] = b0; s2[1] = b1;
  }

  v8f acc[4] = {};
#pragma unroll 2
  for (int it = 0; it < 16; ++it) {
    int cur = it & 1;
    if (it < 15) {
      int k0 = (it + 1) * 32;
      a0 = *(const v8h*)(gA + k0);
      a1 = *(const v8h*)(gA + k0 + 8);
      b0 = *(const v8h*)(gB + k0);
      b1 = *(const v8h*)(gB + k0 + 8);
    }
    __syncthreads();
    if (it < 15) {
      int nxt = cur ^ 1;
      v8h* s = (v8h*)(sA[nxt] + r * 32 + c);
      s[0] = a0; s[1] = a1;
      v8h* s2 = (v8h*)(sB[nxt] + r * 32 + c);
      s2[0] = b0; s2[1] = b1;
    }
    v16h af = lds_a_frag(sA[cur], 32, wave * 16, 0, lane);
#pragma unroll
    for (int tn = 0; tn < 4; ++tn) {
      v16h bf = lds_b_frag(sB[cur], 32, tn * 16, 0, lane);
      acc[tn] = wmma16(af, bf, acc[tn]);
    }
  }

  int hi = lane >> 4, col = lane & 15;
  int i_base = m0 + wave * 16;
#pragma unroll
  for (int tn = 0; tn < 4; ++tn) {
    int j = n0 + tn * 16 + col;
    float bias = out_b[j];
#pragma unroll
    for (int g = 0; g < 8; ++g) {
      int i = i_base + g + 8 * hi;
      out[(size_t)i * HH + j] = acc[tn][g] + bias;
    }
  }
}

// ---------------- Host launcher ----------------

extern "C" void kernel_launch(void* const* d_in, const int* in_sizes, int n_in,
                              void* d_out, int out_size, void* d_ws,
                              size_t ws_size, hipStream_t stream) {
  const float* seqs      = (const float*)d_in[0];
  // d_in[1] (mask) unused: it is tril, applied analytically in attn_kernel
  const int* action_ts   = (const int*)d_in[2];
  const float* qkvu_proj = (const float*)d_in[3];
  const float* out_w     = (const float*)d_in[4];
  const float* out_b     = (const float*)d_in[5];
  const float* ts_w      = (const float*)d_in[6];
  const float* pos_w     = (const float*)d_in[7];
  float* out = (float*)d_out;

  char* p = (char*)d_ws;
  half_t* xh  = (half_t*)p; p += (size_t)ROWS * HH * 2;         // ln(seqs) f16
  half_t* whT = (half_t*)p; p += (size_t)HH * QKVU_N * 2;       // [n][k] f16
  half_t* owh = (half_t*)p; p += (size_t)HH * HH * 2;           // out_w f16
  half_t* qh  = (half_t*)p; p += (size_t)BB * NHD * NN * HD * 2;
  half_t* kh  = (half_t*)p; p += (size_t)BB * NHD * NN * HD * 2;
  half_t* vth = (half_t*)p; p += (size_t)BB * NHD * NN * HD * 2; // [bh][d][n]
  float* u_ws  = (float*)p; p += (size_t)ROWS * HH * 4;
  float* av_ws = (float*)p; p += (size_t)ROWS * HH * 4;
  half_t* yh  = (half_t*)p; p += (size_t)ROWS * HH * 2;

  int nw = HH * QKVU_N;
  w_transpose_kernel<<<nw / 256, 256, 0, stream>>>(qkvu_proj, whT);
  int no = HH * HH;
  f32_to_f16_kernel<<<(no + 255) / 256, 256, 0, stream>>>(out_w, owh, no);

  ln_kernel<<<ROWS, 256, 0, stream>>>(seqs, xh);

  gemm_qkvu_kernel<<<dim3(QKVU_N / 64, ROWS / 64), 128, 0, stream>>>(
      xh, whT, qh, kh, vth, u_ws);

  attn_kernel<<<dim3(BB * NHD, NN / 64), 128, 0, stream>>>(
      qh, kh, vth, action_ts, ts_w, pos_w, av_ws);

  lnu_kernel<<<ROWS, 256, 0, stream>>>(av_ws, u_ws, yh);

  gemm_out_kernel<<<dim3(HH / 64, ROWS / 64), 128, 0, stream>>>(yh, owh, out_b,
                                                                out);
}
